// AE_spikes_latency_60902636257521
// MI455X (gfx1250) — matrix operand order repaired and checked by reading, not compiled, not run on hardware
//
#include <hip/hip_runtime.h>
#include <hip/hip_bf16.h>

typedef _Float16 v8h  __attribute__((ext_vector_type(8)));
typedef _Float16 v16h __attribute__((ext_vector_type(16)));
typedef float    v4f  __attribute__((ext_vector_type(4)));
typedef float    v8f  __attribute__((ext_vector_type(8)));

#define BATCH   65536
#define IN_DIM  784
#define HID     128

__device__ __forceinline__ v8h zero8h() {
    v8h z;
#pragma unroll
    for (int i = 0; i < 8; ++i) z[i] = (_Float16)0.0f;
    return z;
}

__device__ __forceinline__ v16h pack16(v8h lo, v8h hi) {
    v16h a;
#pragma unroll
    for (int i = 0; i < 8; ++i) { a[i] = lo[i]; a[i + 8] = hi[i]; }
    return a;
}

// Quantize 8 f32 -> integer spike counts 0..255, exact in f16.
__device__ __forceinline__ v8h quant8(v4f x0, v4f x1, float inv_h) {
    v8h q;
#pragma unroll
    for (int i = 0; i < 4; ++i) {
        q[i]     = (_Float16)fminf(fmaxf(floorf(x0[i] * inv_h), 0.0f), 255.0f);
        q[i + 4] = (_Float16)fminf(fmaxf(floorf(x1[i] * inv_h), 0.0f), 255.0f);
    }
    return q;
}

// ---------------------------------------------------------------------------
// Core GEMM body, software-pipelined over K: step kb's loads are issued
// before the WMMA chain consuming step kb-32's fragments, with ping-pong
// fragment sets so the allocator cannot merge them (kills the per-tile
// s_wait_loadcnt 0 stalls). NT (output tiles) is compile-time => branchless.
//   A fragment (16x32, M-major): lane L row L&15, chunks at k = 8*(L>>4), +16.
//   B fragment (32x16, K-major): lane L col L&15, chunks at k = 8*(L>>4), +16.
// B tile displacement t*16*KP is a literal folded into the load immediate.
// ---------------------------------------------------------------------------
template <int K, int N, int NT, bool A_F32, bool QUANT_OUT>
__device__ __forceinline__ void gemm_core(
    const void* __restrict__ Ain, const _Float16* __restrict__ Wh,
    const float* __restrict__ bias, void* __restrict__ Out,
    int rowBase, int nBase, float scale, float ih_in, float ih_out) {
    constexpr int KP    = (K + 31) & ~31;  // B rows padded to this pitch
    constexpr int KFULL = K & ~31;
    static_assert(K == KFULL || K - KFULL == 16, "tail handling assumes K%32 in {0,16}");
    static_assert(KFULL >= 32, "need at least one full K step");

    const int lane = threadIdx.x & 31;
    const int r    = lane & 15;
    const int g    = lane >> 4;

    const uint32_t aOff = (uint32_t)(rowBase + r) * (uint32_t)K + (uint32_t)(g << 3);
    const uint32_t bOff = (uint32_t)(nBase + r) * (uint32_t)KP + (uint32_t)(g << 3);
    const _Float16* __restrict__ Ah = (const _Float16*)Ain;
    const float* __restrict__ Af    = (const float*)Ain;

    auto loadA = [&](int kb) -> v16h {
        if constexpr (A_F32) {
            v4f x0 = __builtin_nontemporal_load((const v4f*)(Af + aOff + kb));
            v4f x1 = __builtin_nontemporal_load((const v4f*)(Af + aOff + kb + 4));
            v4f x2 = __builtin_nontemporal_load((const v4f*)(Af + aOff + kb + 16));
            v4f x3 = __builtin_nontemporal_load((const v4f*)(Af + aOff + kb + 20));
            return pack16(quant8(x0, x1, ih_in), quant8(x2, x3, ih_in));
        } else {
            return pack16(*(const v8h*)(Ah + aOff + kb),
                          *(const v8h*)(Ah + aOff + kb + 16));
        }
    };
    // Tail K-step A: chunk0 (k in [KFULL, K)) in range for both lane halves,
    // chunk1 (k >= K) entirely out of range -> zeros (compile-time decision).
    auto loadA_tail = [&]() -> v16h {
        if constexpr (A_F32) {
            v4f x0 = __builtin_nontemporal_load((const v4f*)(Af + aOff + KFULL));
            v4f x1 = __builtin_nontemporal_load((const v4f*)(Af + aOff + KFULL + 4));
            return pack16(quant8(x0, x1, ih_in), zero8h());
        } else {
            return pack16(*(const v8h*)(Ah + aOff + KFULL), zero8h());
        }
    };
    auto loadB = [&](int kb, v16h (&bf)[NT]) {
#pragma unroll
        for (int t = 0; t < NT; ++t)
            bf[t] = pack16(*(const v8h*)(Wh + bOff + kb + t * 16 * KP),
                           *(const v8h*)(Wh + bOff + kb + t * 16 * KP + 16));
    };

    v8f acc[NT];
#pragma unroll
    for (int t = 0; t < NT; ++t)
#pragma unroll
        for (int v = 0; v < 8; ++v) acc[t][v] = 0.0f;

    // --- pipeline prologue: fragments for step 0 ---
    v16h a_cur = loadA(0);
    v16h b_cur[NT];
    loadB(0, b_cur);

    // --- steady state: issue step kb's loads, then MMA step kb-32 ---
#pragma unroll 2
    for (int kb = 32; kb < KFULL; kb += 32) {
        v16h a_nxt = loadA(kb);
        v16h b_nxt[NT];
        loadB(kb, b_nxt);
#pragma unroll
        for (int t = 0; t < NT; ++t)
            acc[t] = __builtin_amdgcn_wmma_f32_16x16x32_f16(
                false, a_cur, false, b_cur[t], (short)0, acc[t], false, false);
        a_cur = a_nxt;
#pragma unroll
        for (int t = 0; t < NT; ++t) b_cur[t] = b_nxt[t];
    }

    // --- drain (+ zero-padded tail step when K%32 == 16) ---
    if constexpr (KFULL != K) {
        v16h a_tail = loadA_tail();
        v16h b_tail[NT];
        loadB(KFULL, b_tail);  // B rows zero-padded to KP: no guard needed
#pragma unroll
        for (int t = 0; t < NT; ++t)
            acc[t] = __builtin_amdgcn_wmma_f32_16x16x32_f16(
                false, a_cur, false, b_cur[t], (short)0, acc[t], false, false);
#pragma unroll
        for (int t = 0; t < NT; ++t)
            acc[t] = __builtin_amdgcn_wmma_f32_16x16x32_f16(
                false, a_tail, false, b_tail[t], (short)0, acc[t], false, false);
    } else {
#pragma unroll
        for (int t = 0; t < NT; ++t)
            acc[t] = __builtin_amdgcn_wmma_f32_16x16x32_f16(
                false, a_cur, false, b_cur[t], (short)0, acc[t], false, false);
    }

    // Epilogue: C layout M = v + 8*(lane>>4), N = lane&15.
    const int mOff = g << 3;
#pragma unroll
    for (int t = 0; t < NT; ++t) {
        const int n0   = nBase + t * 16;
        const float bv = bias[n0 + r];
#pragma unroll
        for (int v = 0; v < 8; ++v) {
            const uint32_t row = (uint32_t)(rowBase + mOff + v);
            float val = fminf(fmaxf(acc[t][v] * scale + bv, 0.0f), 1000.0f);
            if constexpr (QUANT_OUT) {
                float q = fminf(floorf(val * ih_out), 255.0f);
                ((_Float16*)Out)[row * (uint32_t)N + n0 + r] = (_Float16)q;
            } else {
                __builtin_nontemporal_store(val, (float*)Out + (size_t)row * N + n0 + r);
            }
        }
    }
}

// Block: 256 threads = 8 waves; wave w owns rows [bx*128+16w, +16),
// cols [by*128, +16*NT).
template <int K, int N, bool A_F32, bool QUANT_OUT>
__global__ __launch_bounds__(256) void spike_gemm(
    const void* __restrict__ Ain, const _Float16* __restrict__ Wh,
    const float* __restrict__ bias, void* __restrict__ Out,
    float scale, float ih_in, float ih_out) {
    const int wave    = threadIdx.x >> 5;
    const int rowBase = blockIdx.x * 128 + wave * 16;
    const int nBase   = blockIdx.y * 128;
    if constexpr (N % 128 == 0) {
        gemm_core<K, N, 8, A_F32, QUANT_OUT>(Ain, Wh, bias, Out, rowBase, nBase,
                                             scale, ih_in, ih_out);
    } else {
        static_assert((N % 128) % 16 == 0, "N must be a multiple of 16");
        if (nBase + 128 <= N)
            gemm_core<K, N, 8, A_F32, QUANT_OUT>(Ain, Wh, bias, Out, rowBase, nBase,
                                                 scale, ih_in, ih_out);
        else
            gemm_core<K, N, (N % 128) / 16, A_F32, QUANT_OUT>(Ain, Wh, bias, Out,
                                                              rowBase, nBase,
                                                              scale, ih_in, ih_out);
    }
}

// f32 -> f16 weight conversion with optional zero-padded K pitch.
__global__ __launch_bounds__(256) void cvt_pad_f16(const float* __restrict__ src,
                                                   _Float16* __restrict__ dst,
                                                   int rows, int ks, int kd) {
    int i = blockIdx.x * blockDim.x + threadIdx.x;
    if (i < rows * kd) {
        int rr = i / kd, k = i - rr * kd;
        dst[i] = (k < ks) ? (_Float16)src[rr * ks + k] : (_Float16)0.0f;
    }
}

extern "C" void kernel_launch(void* const* d_in, const int* in_sizes, int n_in,
                              void* d_out, int out_size, void* d_ws, size_t ws_size,
                              hipStream_t stream) {
    const float* features = (const float*)d_in[0];
    const float* W1 = (const float*)d_in[1];
    const float* b1 = (const float*)d_in[2];
    const float* W2 = (const float*)d_in[3];
    const float* b2 = (const float*)d_in[4];
    const float* W3 = (const float*)d_in[5];
    const float* b3 = (const float*)d_in[6];
    const float* W4 = (const float*)d_in[7];
    const float* b4 = (const float*)d_in[8];
    // Uniform linspace bins -> closed-form spike heights.
    const float h0  = 1.0f / 255.0f;
    const float h   = 20.0f / 255.0f;
    const float ih0 = 255.0f;
    const float ih  = 255.0f / 20.0f;  // 12.75 exact

    constexpr int KP1 = (IN_DIM + 31) & ~31;  // 800: W1 rows zero-padded

    char* ws = (char*)d_ws;
    size_t off = 0;
    auto carve = [&](size_t bytes) -> void* {
        void* p = ws + off;
        off = (off + bytes + 255) & ~(size_t)255;
        return p;
    };
    _Float16* W1h = (_Float16*)carve((size_t)HID * KP1 * 2);
    _Float16* W2h = (_Float16*)carve((size_t)HID * HID * 2);
    _Float16* W3h = (_Float16*)carve((size_t)HID * HID * 2);
    _Float16* W4h = (_Float16*)carve((size_t)IN_DIM * HID * 2);
    _Float16* qx1 = (_Float16*)carve((size_t)BATCH * HID * 2);
    _Float16* qx2 = (_Float16*)carve((size_t)BATCH * HID * 2);
    _Float16* qx3 = (_Float16*)carve((size_t)BATCH * HID * 2);
    (void)ws_size; (void)in_sizes; (void)n_in; (void)out_size;

    cvt_pad_f16<<<(HID * KP1 + 255) / 256, 256, 0, stream>>>(W1, W1h, HID, IN_DIM, KP1);
    cvt_pad_f16<<<(HID * HID + 255) / 256, 256, 0, stream>>>(W2, W2h, HID, HID, HID);
    cvt_pad_f16<<<(HID * HID + 255) / 256, 256, 0, stream>>>(W3, W3h, HID, HID, HID);
    cvt_pad_f16<<<(IN_DIM * HID + 255) / 256, 256, 0, stream>>>(W4, W4h, IN_DIM, HID, HID);

    dim3 blk(256);
    spike_gemm<IN_DIM, HID, true, true>
        <<<dim3(BATCH / 128, 1), blk, 0, stream>>>(features, W1h, b1, qx1, h0, ih0, ih);
    spike_gemm<HID, HID, false, true>
        <<<dim3(BATCH / 128, 1), blk, 0, stream>>>(qx1, W2h, b2, qx2, h, 0.0f, ih);
    spike_gemm<HID, HID, false, true>
        <<<dim3(BATCH / 128, 1), blk, 0, stream>>>(qx2, W3h, b3, qx3, h, 0.0f, ih);
    spike_gemm<HID, IN_DIM, false, false>
        <<<dim3(BATCH / 128, (IN_DIM + 127) / 128), blk, 0, stream>>>(qx3, W4h, b4,
                                                                      d_out, h, 0.0f, 0.0f);
}
